// Seq2SeqInterAttn_53188874993876
// MI455X (gfx1250) — matrix operand order repaired and checked by reading, not compiled, not run on hardware
//
#include <hip/hip_runtime.h>
#include <math.h>

typedef unsigned short u16;
typedef unsigned int   u32;
typedef __attribute__((ext_vector_type(16))) __bf16 v16bf;
typedef __attribute__((ext_vector_type(8)))  float  v8f;

// ---------------- constants -------------------------------------------------
// B=64, T=512, F=128, H=256, OUT=3, 2 layers, Tenc=Tdec=256, ATTN=H
#define BB   64
#define TT   512
#define FF   128
#define HH   256
#define GG   1024      // 4*H
#define TE   256
#define AA   256

__device__ __forceinline__ u16 f2bf(float f) {
    u32 u = __float_as_uint(f);
    u32 r = (u + 0x7FFFu + ((u >> 16) & 1u)) >> 16;
    return (u16)r;
}
__device__ __forceinline__ float sigm(float x) { return 1.f / (1.f + __expf(-x)); }

// ---------------- WMMA fragment helpers -------------------------------------
// A (16x32 bf16, row-major, stride lda): lanes 0-15 row M=lane hold K
// {0..7,16..23}; lanes 16-31 hold K {8..15,24..31}.
// B from weights W[N][K] row-major (operand B(k,n)=W[n][k]): lanes 0-15 col
// N=lane hold K {0..7,16..23}; lanes 16-31 hold K {8..15,24..31}.
union V16 { uint4 q[2]; v16bf v; };

__device__ __forceinline__ v16bf load_fragA(const u16* __restrict__ A, int lda) {
    int lane = threadIdx.x & 31;
    int half = lane >> 4, l15 = lane & 15;
    V16 a;
    const u16* ap = A + (size_t)l15 * lda + half * 8;
    a.q[0] = *(const uint4*)(ap);
    a.q[1] = *(const uint4*)(ap + 16);
    return a.v;
}
__device__ __forceinline__ v16bf load_fragB(const u16* __restrict__ B, int ldb) {
    int lane = threadIdx.x & 31;
    int half = lane >> 4, l15 = lane & 15;
    V16 b;
    const u16* bp = B + (size_t)l15 * ldb + half * 8;
    b.q[0] = *(const uint4*)(bp);
    b.q[1] = *(const uint4*)(bp + 16);
    return b.v;
}
__device__ __forceinline__ v8f wmma_acc(v16bf a, v16bf b, v8f acc) {
    return __builtin_amdgcn_wmma_f32_16x16x32_bf16(false, a, false, b,
                                                   (short)0, acc, false, false);
}

// ---------------- LSTM cell step --------------------------------------------
// grid: 16 WGs x 256 thr (8 waves). WG k owns hidden units [16k,16k+16).
// Wave w: gate g = w&3, M tiles m0 = (w>>2)*2 and m0+1 (B fragment reused in
// registers across the two WMMAs). Activations staged in LDS (1x global read).
__global__ void __launch_bounds__(256)
k_lstm_cell(const u16* __restrict__ xin, int x_ld, int Kin,
            const u16* __restrict__ Wih,          // [1024][Kin] bf16
            const u16* __restrict__ Whh,          // [1024][256] bf16
            const float* __restrict__ bias,       // [1024] f32
            const u16* __restrict__ hin,          // [64][256] bf16
            float* __restrict__ c_state,          // [64][256] f32 (in/out)
            u16* __restrict__ hout,               // [64][256] bf16
            float* __restrict__ out_f32, int f32_ld,   // optional
            u16* __restrict__ out_bf2, int bf2_ld)     // optional
{
    __shared__ u16   sX[64 * 256];        // 32 KB (xin, packed ld=Kin)
    __shared__ u16   sH[64 * 256];        // 32 KB (hin)
    __shared__ float gates[4][64][16];    // 16 KB
    int tid  = threadIdx.x;
    int wave = tid >> 5;
    int lane = tid & 31;
    int half = lane >> 4, l15 = lane & 15;
    int j0   = blockIdx.x * 16;

    // ---- stage activations into LDS (16B vector copies) ----
    for (int i = tid * 8; i < 64 * Kin; i += 256 * 8)
        *(uint4*)&sX[i] = *(const uint4*)&xin[(size_t)(i / Kin) * x_ld + (i % Kin)];
    for (int i = tid * 8; i < 64 * HH; i += 256 * 8)
        *(uint4*)&sH[i] = *(const uint4*)&hin[i];
    __syncthreads();

    int g  = wave & 3;                 // gate (i,f,g,o)
    int m0 = (wave >> 2) * 2;          // batch tile pair {0,1} or {2,3}
    int nc = g * HH + j0;              // gate column base
    v8f acc0 = {}, acc1 = {};

    const u16* Wr = Wih + (size_t)nc * Kin;
    for (int k = 0; k < Kin; k += 32) {
        v16bf bf_ = load_fragB(Wr + k, Kin);
        v16bf a0  = load_fragA(&sX[(16 * m0) * Kin + k], Kin);
        v16bf a1  = load_fragA(&sX[(16 * m0 + 16) * Kin + k], Kin);
        acc0 = wmma_acc(a0, bf_, acc0);
        acc1 = wmma_acc(a1, bf_, acc1);
    }
    const u16* Wr2 = Whh + (size_t)nc * HH;
    for (int k = 0; k < HH; k += 32) {
        v16bf bf_ = load_fragB(Wr2 + k, HH);
        v16bf a0  = load_fragA(&sH[(16 * m0) * HH + k], HH);
        v16bf a1  = load_fragA(&sH[(16 * m0 + 16) * HH + k], HH);
        acc0 = wmma_acc(a0, bf_, acc0);
        acc1 = wmma_acc(a1, bf_, acc1);
    }
    #pragma unroll
    for (int r = 0; r < 8; ++r) {
        gates[g][16 * m0 + half * 8 + r][l15]       = acc0[r];
        gates[g][16 * (m0 + 1) + half * 8 + r][l15] = acc1[r];
    }
    __syncthreads();

    // ---- cell update (fp32) for the 16 owned units ----
    for (int p = tid; p < BB * 16; p += 256) {
        int b = p >> 4, u = p & 15;
        int j = j0 + u;
        float zi = gates[0][b][u] + bias[j];
        float zf = gates[1][b][u] + bias[HH + j];
        float zg = gates[2][b][u] + bias[2 * HH + j];
        float zo = gates[3][b][u] + bias[3 * HH + j];
        float c_old = c_state[b * HH + j];
        float c_new = sigm(zf) * c_old + sigm(zi) * tanhf(zg);
        float h_new = sigm(zo) * tanhf(c_new);
        c_state[b * HH + j] = c_new;
        hout[b * HH + j]    = f2bf(h_new);
        if (out_f32) out_f32[(size_t)b * f32_ld + j] = h_new;
        if (out_bf2) out_bf2[(size_t)b * bf2_ld + j] = f2bf(h_new);
    }
}

// ---------------- generic small GEMM: out = act(A @ W^T + bias) -------------
__global__ void __launch_bounds__(256)
k_gemm(const u16* __restrict__ A, int lda,
       const u16* __restrict__ W,             // [N][K] bf16
       const float* __restrict__ bias,        // or null
       int M, int N, int K,
       float* __restrict__ outf, int ldo,
       int act)                               // 0 none, 1 tanh
{
    int tid  = threadIdx.x;
    int wave = tid >> 5, lane = tid & 31;
    int half = lane >> 4, l15 = lane & 15;
    int mt = M >> 4, nt = N >> 4, total = mt * nt;
    int gw = blockIdx.x * 8 + wave;
    int nw = gridDim.x * 8;
    for (int tt = gw; tt < total; tt += nw) {
        int m = tt % mt, n = tt / mt;
        v8f acc = {};
        const u16* Ar = A + (size_t)(16 * m) * lda;
        const u16* Wr = W + (size_t)(16 * n) * K;
        for (int k = 0; k < K; k += 32)
            acc = wmma_acc(load_fragA(Ar + k, lda), load_fragB(Wr + k, K), acc);
        #pragma unroll
        for (int r = 0; r < 8; ++r) {
            int row = 16 * m + half * 8 + r;
            int col = 16 * n + l15;
            float v = acc[r];
            if (bias) v += bias[col];
            if (act == 1) v = tanhf(v);
            outf[(size_t)row * ldo + col] = v;
        }
    }
}

// ---------------- attention scores ------------------------------------------
// scores[b,t] = sum_a v[a]*tanh( (enc_out[b,t,:]@We[a,:]) + dp[b,a] )
// grid: 512 WGs x 32 rows of flattened (b*256+t). Wave w: m = w&1,
// n in {w>>1, +4, +8, +12}; A fragment reused across the 4 N tiles.
__global__ void __launch_bounds__(256)
k_scores(const u16* __restrict__ encA,   // [16384][256] bf16
         const u16* __restrict__ WeB,    // [256][256] bf16
         const float* __restrict__ dp,   // [64][256]
         const float* __restrict__ vvec, // [256]
         float* __restrict__ scores)     // [16384]
{
    __shared__ float z[32][AA];          // 32 KB
    __shared__ float dpv[AA], vv[AA];
    __shared__ float part[32][8];
    int tid = threadIdx.x, wave = tid >> 5, lane = tid & 31;
    int half = lane >> 4, l15 = lane & 15;
    int r0 = blockIdx.x * 32;
    int b  = r0 >> 8;
    dpv[tid] = dp[b * AA + tid];
    vv[tid]  = vvec[tid];

    int m  = wave & 1;
    int n0 = wave >> 1;
    v8f acc[4] = {};
    const u16* Ar = encA + (size_t)(r0 + 16 * m) * HH;
    for (int k = 0; k < HH; k += 32) {
        v16bf af = load_fragA(Ar + k, HH);
        #pragma unroll
        for (int j = 0; j < 4; ++j) {
            const u16* Wr = WeB + (size_t)(16 * (n0 + 4 * j)) * HH;
            acc[j] = wmma_acc(af, load_fragB(Wr + k, HH), acc[j]);
        }
    }
    #pragma unroll
    for (int j = 0; j < 4; ++j)
        #pragma unroll
        for (int r = 0; r < 8; ++r)
            z[16 * m + half * 8 + r][16 * (n0 + 4 * j) + l15] = acc[j][r];
    __syncthreads();

    int rr = tid >> 3, seg = tid & 7;
    float s = 0.f;
    for (int a = seg * 32; a < seg * 32 + 32; ++a)
        s += vv[a] * tanhf(z[rr][a] + dpv[a]);
    part[rr][seg] = s;
    __syncthreads();
    if (tid < 32) {
        float t2 = 0.f;
        #pragma unroll
        for (int q = 0; q < 8; ++q) t2 += part[tid][q];
        scores[r0 + tid] = t2;
    }
}

// ---------------- softmax over t + context, writes cat[:,256:512] ----------
__global__ void __launch_bounds__(256)
k_softmax_ctx(const float* __restrict__ scores,   // [64][256]
              const float* __restrict__ enc_out,  // [64][256][256] f32
              u16* __restrict__ cat)              // [64][512] bf16
{
    __shared__ float e[TE];
    __shared__ float red[TE];
    int b = blockIdx.x, tid = threadIdx.x;
    float sc = scores[b * TE + tid];
    red[tid] = sc; __syncthreads();
    for (int s = 128; s > 0; s >>= 1) {
        if (tid < s) red[tid] = fmaxf(red[tid], red[tid + s]);
        __syncthreads();
    }
    float mx = red[0]; __syncthreads();
    float ex = __expf(sc - mx);
    e[tid] = ex; red[tid] = ex; __syncthreads();
    for (int s = 128; s > 0; s >>= 1) {
        if (tid < s) red[tid] += red[tid + s];
        __syncthreads();
    }
    float inv = 1.f / red[0];
    __syncthreads();
    const float* eo = enc_out + (size_t)b * TE * HH + tid;
    float acc = 0.f;
    for (int t = 0; t < TE; ++t) acc += e[t] * eo[(size_t)t * HH];
    cat[b * (2 * HH) + HH + tid] = f2bf(acc * inv);
}

// ---------------- small helpers ---------------------------------------------
__global__ void k_cvt(const float* __restrict__ s, u16* __restrict__ d, int n) {
    int i = blockIdx.x * 256 + threadIdx.x;
    if (i < n) d[i] = f2bf(s[i]);
}
__global__ void k_zero16(u16* d, int n) {
    int i = blockIdx.x * 256 + threadIdx.x;
    if (i < n) d[i] = 0;
}
__global__ void k_zero32(float* d, int n) {
    int i = blockIdx.x * 256 + threadIdx.x;
    if (i < n) d[i] = 0.f;
}
__global__ void k_cat_dec(const u16* __restrict__ h, u16* __restrict__ cat) {
    int i = blockIdx.x * 256 + threadIdx.x;     // 16384
    int b = i >> 8, j = i & 255;
    cat[b * (2 * HH) + j] = h[i];
}
__global__ void k_h1(const float* __restrict__ fused, const float* __restrict__ Wh1,
                     const float* __restrict__ bh1, float* __restrict__ h1) {
    int o = blockIdx.x * 256 + threadIdx.x;     // 16384
    int b = o >> 8, j = o & 255;
    const float* f = fused + b * HH;
    const float* w = Wh1 + j * HH;
    float a = bh1[j];
    for (int k = 0; k < HH; ++k) a += f[k] * w[k];
    h1[o] = fmaxf(a, 0.f);
}
__global__ void k_logits(const float* __restrict__ h1, const float* __restrict__ Wh2,
                         const float* __restrict__ bh2, float* __restrict__ out) {
    int i = threadIdx.x;
    if (i >= BB * 3) return;
    int b = i / 3, o = i % 3;
    const float* h = h1 + b * HH;
    const float* w = Wh2 + o * HH;
    float a = bh2[o];
    for (int k = 0; k < HH; ++k) a += h[k] * w[k];
    out[b * 3 + o] = a;
}

// ---------------- host ------------------------------------------------------
extern "C" void kernel_launch(void* const* d_in, const int* in_sizes, int n_in,
                              void* d_out, int out_size, void* d_ws, size_t ws_size,
                              hipStream_t stream) {
    const float *x, *eWih0, *eWhh0, *eb0, *eWih1, *eWhh1, *eb1;
    const float *dWih0, *dWhh0, *db0, *dWih1, *dWhh1, *db1;
    const float *We, *Wd, *vvec, *Wf, *bfv, *Wh1, *bh1, *Wh2, *bh2;
    const float* P[22];
    for (int i = 0; i < 22 && i < n_in; ++i) P[i] = (const float*)d_in[i];

    if (in_sizes[0] == BB * TT * FF) {
        // insertion order: x, enc(l0,l1){Wih,Whh,b}, dec..., We,Wd,v,Wf,bf,Wh1,bh1,Wh2,bh2
        x = P[0];
        eWih0 = P[1]; eWhh0 = P[2]; eb0 = P[3]; eWih1 = P[4]; eWhh1 = P[5]; eb1 = P[6];
        dWih0 = P[7]; dWhh0 = P[8]; db0 = P[9]; dWih1 = P[10]; dWhh1 = P[11]; db1 = P[12];
        We = P[13]; Wd = P[14]; vvec = P[15]; Wf = P[16]; bfv = P[17];
        Wh1 = P[18]; bh1 = P[19]; Wh2 = P[20]; bh2 = P[21];
    } else {
        // jax-sorted tree order: Wd,We,Wf,Wh1,Wh2,bf,bh1,bh2, dec{Whh,Wih,b}x2, enc{...}x2, v, x
        Wd = P[0]; We = P[1]; Wf = P[2]; Wh1 = P[3]; Wh2 = P[4];
        bfv = P[5]; bh1 = P[6]; bh2 = P[7];
        dWhh0 = P[8]; dWih0 = P[9]; db0 = P[10]; dWhh1 = P[11]; dWih1 = P[12]; db1 = P[13];
        eWhh0 = P[14]; eWih0 = P[15]; eb0 = P[16]; eWhh1 = P[17]; eWih1 = P[18]; eb1 = P[19];
        vvec = P[20]; x = P[21];
    }

    // ---- workspace carve-up ----
    char* base = (char*)d_ws;
    size_t off = 0;
    auto carve = [&](size_t bytes) -> char* {
        char* p = base + off;
        off = (off + bytes + 255) & ~(size_t)255;
        return p;
    };
    u16*   xbf    = (u16*)carve((size_t)BB * TT * FF * 2);   // 8 MB
    u16*   eW0i   = (u16*)carve((size_t)GG * FF * 2);
    u16*   eW0h   = (u16*)carve((size_t)GG * HH * 2);
    u16*   eW1i   = (u16*)carve((size_t)GG * HH * 2);
    u16*   eW1h   = (u16*)carve((size_t)GG * HH * 2);
    u16*   dW0i   = (u16*)carve((size_t)GG * FF * 2);
    u16*   dW0h   = (u16*)carve((size_t)GG * HH * 2);
    u16*   dW1i   = (u16*)carve((size_t)GG * HH * 2);
    u16*   dW1h   = (u16*)carve((size_t)GG * HH * 2);
    u16*   Webf   = (u16*)carve((size_t)AA * HH * 2);
    u16*   Wdbf   = (u16*)carve((size_t)AA * HH * 2);
    u16*   Wfbf   = (u16*)carve((size_t)HH * 2 * HH * 2);
    u16*   h0b[2]; h0b[0] = (u16*)carve(BB * HH * 2); h0b[1] = (u16*)carve(BB * HH * 2);
    u16*   h1b[2]; h1b[0] = (u16*)carve(BB * HH * 2); h1b[1] = (u16*)carve(BB * HH * 2);
    float* c0     = (float*)carve(BB * HH * 4);
    float* c1     = (float*)carve(BB * HH * 4);
    float* encOut = (float*)carve((size_t)BB * TE * HH * 4);   // 16 MB
    u16*   encBf  = (u16*)carve((size_t)BB * TE * HH * 2);     // 8 MB
    float* dp     = (float*)carve(BB * AA * 4);
    float* scores = (float*)carve(BB * TE * 4);
    u16*   cat    = (u16*)carve(BB * 2 * HH * 2);
    float* fused  = (float*)carve(BB * HH * 4);
    float* h1     = (float*)carve(BB * HH * 4);
    (void)ws_size;

    auto cvt = [&](const float* s, u16* d, int n) {
        k_cvt<<<(n + 255) / 256, 256, 0, stream>>>(s, d, n);
    };
    // weights & input -> bf16
    cvt(x, xbf, BB * TT * FF);
    cvt(eWih0, eW0i, GG * FF);  cvt(eWhh0, eW0h, GG * HH);
    cvt(eWih1, eW1i, GG * HH);  cvt(eWhh1, eW1h, GG * HH);
    cvt(dWih0, dW0i, GG * FF);  cvt(dWhh0, dW0h, GG * HH);
    cvt(dWih1, dW1i, GG * HH);  cvt(dWhh1, dW1h, GG * HH);
    cvt(We, Webf, AA * HH); cvt(Wd, Wdbf, AA * HH); cvt(Wf, Wfbf, HH * 2 * HH);
    // zero initial states (ping buffer 0, cells)
    k_zero16<<<(BB * HH + 255) / 256, 256, 0, stream>>>(h0b[0], BB * HH);
    k_zero16<<<(BB * HH + 255) / 256, 256, 0, stream>>>(h1b[0], BB * HH);
    k_zero32<<<(BB * HH + 255) / 256, 256, 0, stream>>>(c0, BB * HH);
    k_zero32<<<(BB * HH + 255) / 256, 256, 0, stream>>>(c1, BB * HH);

    // ---- encoder: 256 steps ----
    for (int t = 0; t < TE; ++t) {
        int p = t & 1, q = p ^ 1;
        k_lstm_cell<<<16, 256, 0, stream>>>(
            xbf + (size_t)t * FF, TT * FF, FF, eW0i, eW0h, eb0,
            h0b[p], c0, h0b[q], (float*)nullptr, 0, (u16*)nullptr, 0);
        k_lstm_cell<<<16, 256, 0, stream>>>(
            h0b[q], HH, HH, eW1i, eW1h, eb1,
            h1b[p], c1, h1b[q],
            encOut + (size_t)t * HH, TE * HH,       // enc_out[b][t][:]
            encBf + (size_t)t * HH, TE * HH);
    }
    // ---- decoder: 256 steps (states continue from encoder) ----
    for (int t = 0; t < TE; ++t) {
        int s = TE + t;
        int p = s & 1, q = p ^ 1;
        k_lstm_cell<<<16, 256, 0, stream>>>(
            xbf + (size_t)s * FF, TT * FF, FF, dW0i, dW0h, db0,
            h0b[p], c0, h0b[q], (float*)nullptr, 0, (u16*)nullptr, 0);
        k_lstm_cell<<<16, 256, 0, stream>>>(
            h0b[q], HH, HH, dW1i, dW1h, db1,
            h1b[p], c1, h1b[q], (float*)nullptr, 0, (u16*)nullptr, 0);
    }
    u16* decState = h1b[(TE + TE) & 1];   // h after step 511 lives in buf[0]

    // ---- attention (needed once, for fused_seq[-1]) ----
    k_cat_dec<<<BB * HH / 256, 256, 0, stream>>>(decState, cat);
    k_gemm<<<8, 256, 0, stream>>>(decState, HH, Wdbf, nullptr,
                                  BB, AA, HH, dp, AA, 0);
    k_scores<<<(BB * TE) / 32, 256, 0, stream>>>(encBf, Webf, dp, vvec, scores);
    k_softmax_ctx<<<BB, 256, 0, stream>>>(scores, encOut, cat);
    k_gemm<<<8, 256, 0, stream>>>(cat, 2 * HH, Wfbf, bfv,
                                  BB, HH, 2 * HH, fused, HH, 1);

    // ---- head ----
    k_h1<<<BB * HH / 256, 256, 0, stream>>>(fused, Wh1, bh1, h1);
    k_logits<<<1, 256, 0, stream>>>(h1, Wh2, bh2, (float*)d_out);
    (void)out_size;
}
